// FullAttention_42356967473348
// MI455X (gfx1250) — matrix operand order repaired and checked by reading, not compiled, not run on hardware
//
#include <hip/hip_runtime.h>
#include <hip/hip_bf16.h>

// Flash-attention forward (causal, clamp(-50,50) after mask) for
// B=2,H=16,L=S=2048,D=128 fp32, on gfx1250 via v_wmma_f32_16x16x32_bf16.
//
// Compute-bound: ~69 GFLOP dense vs ~6us of HBM traffic at 23.3 TB/s.
// - one f32->bf16 (RNE) conversion of Q/K/V, packed 2-at-a-time, vectorized
//   LDS stores (b128 for Q/K, b64 for transposed V)
// - both GEMMs on 16x16x32 bf16 WMMA with f32 accumulation, softmax in f32
// - K/V double-buffered in LDS; global loads for block kb+1 issue before the
//   WMMA work of block kb (one barrier per iteration)
// - global_prefetch_b8 two blocks ahead; non-temporal output stores

typedef __bf16 bf16_t;
typedef __attribute__((ext_vector_type(8)))  __bf16 v8bf;
typedef __attribute__((ext_vector_type(16))) __bf16 v16bf;
typedef __attribute__((ext_vector_type(8)))  float  v8f;
typedef __attribute__((ext_vector_type(4)))  float  f32x4;

constexpr int kB  = 2;
constexpr int kH  = 16;
constexpr int kL  = 2048;
constexpr int kD  = 128;

constexpr int BR     = 128;          // query rows per workgroup (8 waves x 16)
constexpr int BC     = 32;           // key columns per block
constexpr int NWAVES = 8;
constexpr int KPAD   = kD + 8;       // 136 bf16 -> 272B row stride (16B aligned)
constexpr int VPAD   = BC + 8;       // 40 bf16  -> 80B  row stride (16B aligned)

constexpr float SCALE = 0.08838834764831845f;  // 1/sqrt(128)
constexpr float LOG2E = 1.4426950408889634f;

struct KvBuf {
  bf16_t k [BC * KPAD];                        // K block, row-major [s][d]
  bf16_t vT[kD * VPAD];                        // V block, transposed [d][s]
};

struct alignas(32) LdsT {
  union {
    bf16_t q[BR * KPAD];                       // phase 1: Q tile (34816 B)
    struct {
      KvBuf  buf[2];                           // double-buffered K/V (37888 B)
      bf16_t p[NWAVES * 16 * VPAD];            // per-wave P scratch (10240 B)
    } kv;
  };
};

// Two f32 -> packed bf16x2 (RNE), low 16 bits = a.
__device__ __forceinline__ unsigned pk_bf16(float a, float b) {
  unsigned ua = __builtin_bit_cast(unsigned, a);
  ua += 0x7fffu + ((ua >> 16) & 1u);
  unsigned ub = __builtin_bit_cast(unsigned, b);
  ub += 0x7fffu + ((ub >> 16) & 1u);
  return (ub & 0xffff0000u) | (ua >> 16);
}

__device__ __forceinline__ bf16_t to_bf16(float f) {
  unsigned u = __builtin_bit_cast(unsigned, f);
  u += 0x7fffu + ((u >> 16) & 1u);
  unsigned short s = (unsigned short)(u >> 16);
  return __builtin_bit_cast(bf16_t, s);
}

// Load 16 bf16 as two 16B chunks p[0..7], p[8..15] (both 16B aligned).
__device__ __forceinline__ v16bf load16(const bf16_t* p) {
  v8bf lo = *(const v8bf*)p;
  v8bf hi = *(const v8bf*)(p + 8);
  return __builtin_shufflevector(lo, hi, 0,1,2,3,4,5,6,7,8,9,10,11,12,13,14,15);
}

// Load 16 bf16 split as p[0..7] and p[gap..gap+7] (A-fragment pattern).
__device__ __forceinline__ v16bf load8x2(const bf16_t* p, int gap) {
  v8bf lo = *(const v8bf*)p;
  v8bf hi = *(const v8bf*)(p + gap);
  return __builtin_shufflevector(lo, hi, 0,1,2,3,4,5,6,7,8,9,10,11,12,13,14,15);
}

// Issue raw global loads of one K/V block into registers (no waits forced).
__device__ __forceinline__ void stage_kv(const float* gK, const float* gV,
                                         int kb, int tid,
                                         f32x4 kreg[4], f32x4 vreg[4]) {
  const int rK = tid >> 3;             // 0..31 key row
  const int dK = (tid & 7) * 16;
  const float* ks = gK + (size_t)(kb * BC + rK) * kD + dK;
#pragma unroll
  for (int i = 0; i < 4; ++i) kreg[i] = *(const f32x4*)(ks + 4 * i);
  const int r0  = (tid >> 5) * 4;      // key rows r0..r0+3 (V 4x4 block)
  const int d0v = (tid & 31) * 4;
  const float* vb = gV + (size_t)(kb * BC + r0) * kD + d0v;
#pragma unroll
  for (int j = 0; j < 4; ++j) vreg[j] = *(const f32x4*)(vb + (size_t)j * kD);
}

// Convert staged registers to bf16 and store into an LDS K/V buffer.
__device__ __forceinline__ void commit_kv(KvBuf* buf, int tid,
                                          const f32x4 kreg[4], const f32x4 vreg[4]) {
  const int rK = tid >> 3;
  const int dK = (tid & 7) * 16;
  bf16_t* kd = &buf->k[rK * KPAD + dK];
#pragma unroll
  for (int i = 0; i < 2; ++i) {
    const f32x4 a = kreg[2 * i], b = kreg[2 * i + 1];
    uint4 u;
    u.x = pk_bf16(a.x, a.y);  u.y = pk_bf16(a.z, a.w);
    u.z = pk_bf16(b.x, b.y);  u.w = pk_bf16(b.z, b.w);
    *(uint4*)(kd + 8 * i) = u;                         // ds_store_b128
  }
  const int r0  = (tid >> 5) * 4;
  const int d0v = (tid & 31) * 4;
#pragma unroll
  for (int i = 0; i < 4; ++i) {
    uint2 u;
    u.x = pk_bf16(vreg[0][i], vreg[1][i]);
    u.y = pk_bf16(vreg[2][i], vreg[3][i]);
    *(uint2*)&buf->vT[(d0v + i) * VPAD + r0] = u;      // ds_store_b64
  }
}

__global__ __launch_bounds__(256) void fa_fwd_causal(
    const float* __restrict__ Qg, const float* __restrict__ Kg,
    const float* __restrict__ Vg, float* __restrict__ Og) {
  __shared__ LdsT lds;

  const int rt   = blockIdx.x;        // query row-tile
  const int bh   = blockIdx.y;        // b*H + h
  const int tid  = threadIdx.x;
  const int wave = tid >> 5;
  const int lane = tid & 31;
  const int lrow  = lane & 15;        // N index of C/B layouts, M index of A layout
  const int lhalf = lane >> 4;

  const size_t base = (size_t)bh * kL * kD;
  const float* gQ = Qg + base;
  const float* gK = Kg + base;
  const float* gV = Vg + base;
  float*       gO = Og + base;

  const int rowBase  = rt * BR;
  const int waveRow0 = rowBase + wave * 16;

  // ---------------- Phase 1: Q tile -> LDS bf16, then A-fragments ----------
  {
    const int r  = tid >> 1;              // 0..127
    const int d0 = (tid & 1) * 64;
    const float* src = gQ + (size_t)(rowBase + r) * kD + d0;
    bf16_t* dst = &lds.q[r * KPAD + d0];
#pragma unroll
    for (int i = 0; i < 64; i += 8) {
      f32x4 a = *(const f32x4*)(src + i);
      f32x4 b = *(const f32x4*)(src + i + 4);
      uint4 u;
      u.x = pk_bf16(a.x, a.y);  u.y = pk_bf16(a.z, a.w);
      u.z = pk_bf16(b.x, b.y);  u.w = pk_bf16(b.z, b.w);
      *(uint4*)(dst + i) = u;                  // ds_store_b128
    }
  }
  __syncthreads();

  // A layout (16x32 bf16): lane: M = lane%16, K(e) = (lane/16)*16 + e
  v16bf qa[4];
#pragma unroll
  for (int kc = 0; kc < 4; ++kc) {
    const bf16_t* qp = &lds.q[(wave * 16 + lrow) * KPAD + kc * 32 + lhalf * 16];
    qa[kc] = load8x2(qp, 8);
  }
  __syncthreads();   // before LDS is reused for K/V

  // ---------------- Softmax / output state --------------------------------
  v8f Ov[8];
#pragma unroll
  for (int dc = 0; dc < 8; ++dc)
#pragma unroll
    for (int i = 0; i < 8; ++i) Ov[dc][i] = 0.0f;
  float m_st[8], l_st[8];
#pragma unroll
  for (int r = 0; r < 8; ++r) { m_st[r] = -1e30f; l_st[r] = 0.0f; }

  const int nkb = (rowBase + BR) / BC;   // key blocks up to (and incl.) diagonal

  // Pipeline prologue: block 0 into buffer 0.
  {
    f32x4 k0[4], v0[4];
    stage_kv(gK, gV, 0, tid, k0, v0);
    commit_kv(&lds.kv.buf[0], tid, k0, v0);
  }
  __syncthreads();

  for (int kb = 0; kb < nkb; ++kb) {
    const int par = kb & 1;
    const bool have_next = (kb + 1 < nkb);

    // -------- issue next block's global loads before this block's compute --
    f32x4 kn[4], vn[4];
    if (have_next) stage_kv(gK, gV, kb + 1, tid, kn, vn);
    if (kb + 2 < nkb) {                  // prefetch one more block ahead
      const size_t off = (size_t)((kb + 2) * BC + (tid >> 3)) * kD + (tid & 7) * 16;
      __builtin_prefetch(gK + off, 0, 2);
      __builtin_prefetch(gV + off, 0, 2);
    }

    const bf16_t* kTile = lds.kv.buf[par].k;
    const bf16_t* vTile = lds.kv.buf[par].vT;

    // Wave-uniform skip of fully-future blocks (EXEC stays all-1s for WMMA).
    if (kb * BC < waveRow0 + 16) {
      // -------- S = Q * K^T : 2 column chunks x 4 K-dim chunks ------------
      v8f sc[2];
#pragma unroll
      for (int c = 0; c < 2; ++c) {
        // Preload all 4 B-fragments so ds_loads issue back-to-back.
        // B layout (32x16): lane: N = lane%16 (key row), K(e) = (lane/16)*16+e (d)
        v16bf kf[4];
#pragma unroll
        for (int kc = 0; kc < 4; ++kc)
          kf[kc] = load16(&kTile[(c * 16 + lrow) * KPAD + kc * 32 + lhalf * 16]);
#pragma unroll
        for (int i = 0; i < 8; ++i) sc[c][i] = 0.0f;
#pragma unroll
        for (int kc = 0; kc < 4; ++kc)
          sc[c] = __builtin_amdgcn_wmma_f32_16x16x32_bf16(
              false, qa[kc], false, kf[kc], (short)0, sc[c], false, false);
      }

      // -------- scale, clamp(+/-50), causal mask (masked -> -50), rowmax --
      float rmax[8];
#pragma unroll
      for (int r = 0; r < 8; ++r) {
        const int qrow = waveRow0 + r + 8 * lhalf;
        float mx = -1e30f;
#pragma unroll
        for (int c = 0; c < 2; ++c) {
          const int skey = kb * BC + c * 16 + lrow;
          float s = sc[c][r] * SCALE;
          s = fminf(fmaxf(s, -50.0f), 50.0f);
          s = (skey > qrow) ? -50.0f : s;   // clamp-after-mask => -50, not -inf
          sc[c][r] = s;
          mx = fmaxf(mx, s);
        }
        rmax[r] = mx;
      }
#pragma unroll
      for (int r = 0; r < 8; ++r)
#pragma unroll
        for (int m = 1; m < 16; m <<= 1)
          rmax[r] = fmaxf(rmax[r], __shfl_xor(rmax[r], m, 32));

      // -------- online softmax update -------------------------------------
      float alpha[8];
#pragma unroll
      for (int r = 0; r < 8; ++r) {
        const float mn = fmaxf(m_st[r], rmax[r]);
        alpha[r] = __builtin_exp2f((m_st[r] - mn) * LOG2E);
        m_st[r] = mn;
      }

      bf16_t* pt = &lds.kv.p[wave * 16 * VPAD];
      float rsum[8];
#pragma unroll
      for (int r = 0; r < 8; ++r) {
        rsum[r] = 0.0f;
#pragma unroll
        for (int c = 0; c < 2; ++c) {
          const float p = __builtin_exp2f((sc[c][r] - m_st[r]) * LOG2E);
          rsum[r] += p;
          // C layout (M=r+8*lhalf, N=lrow) -> row-major P scratch
          pt[(r + 8 * lhalf) * VPAD + c * 16 + lrow] = to_bf16(p);
        }
      }
#pragma unroll
      for (int r = 0; r < 8; ++r) {
#pragma unroll
        for (int m = 1; m < 16; m <<= 1)
          rsum[r] += __shfl_xor(rsum[r], m, 32);
        l_st[r] = l_st[r] * alpha[r] + rsum[r];
      }
#pragma unroll
      for (int dc = 0; dc < 8; ++dc)
#pragma unroll
        for (int r = 0; r < 8; ++r) Ov[dc][r] *= alpha[r];

      // -------- O += P * V ------------------------------------------------
      // A-fragment of P (intra-wave LDS round-trip = C->A transpose),
      // then preload all 8 V B-fragments before the WMMA chain.
      const bf16_t* pr = pt + lrow * VPAD + lhalf * 16;
      v16bf pa = load8x2(pr, 8);
      v16bf vf[8];
#pragma unroll
      for (int dc = 0; dc < 8; ++dc)
        vf[dc] = load16(&vTile[(dc * 16 + lrow) * VPAD + lhalf * 16]);
#pragma unroll
      for (int dc = 0; dc < 8; ++dc)
        Ov[dc] = __builtin_amdgcn_wmma_f32_16x16x32_bf16(
            false, pa, false, vf[dc], (short)0, Ov[dc], false, false);
    }

    // -------- commit next block into the other buffer ----------------------
    if (have_next) commit_kv(&lds.kv.buf[par ^ 1], tid, kn, vn);
    __syncthreads();
  }

  // ---------------- normalize + non-temporal store -------------------------
#pragma unroll
  for (int r = 0; r < 8; ++r) {
    const float inv = 1.0f / l_st[r];
    const int qrow = waveRow0 + r + 8 * lhalf;
    float* op = gO + (size_t)qrow * kD + lrow;
#pragma unroll
    for (int dc = 0; dc < 8; ++dc)
      __builtin_nontemporal_store(Ov[dc][r] * inv, op + dc * 16);
  }
}

extern "C" void kernel_launch(void* const* d_in, const int* in_sizes, int n_in,
                              void* d_out, int out_size, void* d_ws, size_t ws_size,
                              hipStream_t stream) {
  const float* Q = (const float*)d_in[0];
  const float* K = (const float*)d_in[1];
  const float* V = (const float*)d_in[2];
  float*       O = (float*)d_out;
  (void)in_sizes; (void)n_in; (void)out_size; (void)d_ws; (void)ws_size;

  dim3 grid(kL / BR, kB * kH);   // (16, 32)
  fa_fwd_causal<<<grid, 256, 0, stream>>>(Q, K, V, O);
}
